// M2BertLFM_81389630259298
// MI455X (gfx1250) — matrix operand order repaired and checked by reading, not compiled, not run on hardware
//
#include <hip/hip_runtime.h>

typedef __bf16 bf16;
typedef __bf16 v16bf __attribute__((ext_vector_type(16)));
typedef float  v8f   __attribute__((ext_vector_type(8)));

// ---------------- model constants ----------------
#define BB 2
#define SS 1024
#define HH 768
#define NHh 12
#define NKVh 4
#define HDh 64
#define VV 50000
#define KK 3
#define NCc 10
#define NAa 6
#define MT (BB*SS)          // 2048 tokens

// ---------------- GEMM tiling ----------------
#define BM 128
#define BN 64
#define BKT 64              // K tile; all K dims are multiples of 64
#define APITCH (BKT + 8)    // 72 elems = 144B rows, 16B aligned
#define BPITCH (BKT + 8)

// Use CDNA5 async global->LDS copies (ASYNCcnt) for the tile pipeline.
#define USE_ASYNC_LDS 1

#if USE_ASYNC_LDS
#define ASYNC_CP_B128(dst, src)                                              \
    asm volatile("global_load_async_to_lds_b128 %0, %1, off"                 \
                 :: "v"((unsigned)(uintptr_t)(dst)),                         \
                    "v"((unsigned long long)(uintptr_t)(src))                \
                 : "memory")
#define ASYNC_WAIT() asm volatile("s_wait_asynccnt 0x0" ::: "memory")
#else
#define ASYNC_CP_B128(dst, src)                                              \
    (*reinterpret_cast<uint4*>(dst) = *reinterpret_cast<const uint4*>(src))
#define ASYNC_WAIT()
#endif

// flags: bit0 = accumulate into C (residual add).
// B operand is ALWAYS stored transposed: Bt[n*ldb + k] (K-contiguous rows).
__global__ __launch_bounds__(256) void gemm_bf16_wmma(
    const bf16* __restrict__ A, const bf16* __restrict__ Bt, float* __restrict__ C,
    const float* __restrict__ bias,
    int M, int N, int K, int lda, int ldb, int ldc,
    long sA1, long sA2, long sB1, long sB2, long sC1, long sC2, int nz2,
    int flags, float scale, int shift, int seg)
{
    __shared__ bf16 As[2][BM][APITCH];
    __shared__ bf16 Bs[2][BN][BPITCH];

    const int tid  = threadIdx.x;
    const int lane = tid & 31;
    const int wave = tid >> 5;
    const int wm0  = (wave & 3) * 32;   // 4 waves along M, each 32 rows
    const int wn0  = (wave >> 2) * 32;  // 2 waves along N, each 32 cols
    const int hl   = lane >> 4;
    const int l16  = lane & 15;

    const int z  = blockIdx.z;
    const int z2 = z % nz2, z1 = z / nz2;
    const bf16* Ag = A  + (size_t)z1 * sA1 + (size_t)z2 * sA2;
    const bf16* Bg = Bt + (size_t)z1 * sB1 + (size_t)z2 * sB2;
    float*      Cg = C  + (size_t)z1 * sC1 + (size_t)z2 * sC2;

    const int mBase = blockIdx.y * BM;
    const int nBase = blockIdx.x * BN;
    const bool fastA = (shift == 0) && (seg == 0) && (mBase + BM <= M);

    // ---- double-buffered tile loader (async global->LDS on gfx1250) ----
    auto load_tiles = [&](int kt, int bsel) {
        const int k0 = kt * BKT;
        // A tile: 128x64, 16B per op -> 1024 ops (4 per thread)
        if (fastA) {
#pragma unroll
            for (int u = 0; u < (BM * BKT) / 8 / 256; ++u) {
                int i = tid + u * 256;
                int r = i >> 3;
                int c = (i & 7) << 3;
                ASYNC_CP_B128(&As[bsel][r][c], Ag + (size_t)(mBase + r) * lda + (k0 + c));
            }
        } else {
#pragma unroll
            for (int u = 0; u < (BM * BKT) / 8 / 256; ++u) {
                int i = tid + u * 256;
                int r = i >> 3;
                int c = (i & 7) << 3;
                int gm = mBase + r;
                int src = gm + shift;
                bool valid = (gm < M);
                if (seg) { int s0 = (gm / seg) * seg; valid = valid && (src >= s0) && (src < s0 + seg); }
                else     { valid = valid && (src >= 0) && (src < M); }
                bf16* dst = &As[bsel][r][c];
                if (valid) {
                    ASYNC_CP_B128(dst, Ag + (size_t)src * lda + (k0 + c));
                } else {
                    *reinterpret_cast<uint4*>(dst) = make_uint4(0u, 0u, 0u, 0u);
                }
            }
        }
        // B tile: 64 rows (N) x 64 (K) -> 512 ops (2 per thread)
#pragma unroll
        for (int u = 0; u < (BN * BKT) / 8 / 256; ++u) {
            int i = tid + u * 256;
            int n = i >> 3;
            int c = (i & 7) << 3;
            int gn = nBase + n;
            bf16* dst = &Bs[bsel][n][c];
            if (gn < N) {
                ASYNC_CP_B128(dst, Bg + (size_t)gn * ldb + (k0 + c));
            } else {
                *reinterpret_cast<uint4*>(dst) = make_uint4(0u, 0u, 0u, 0u);
            }
        }
    };

    v8f acc[2][2];
    const v8f vzero = {0.f,0.f,0.f,0.f,0.f,0.f,0.f,0.f};
#pragma unroll
    for (int i = 0; i < 2; ++i)
#pragma unroll
        for (int j = 0; j < 2; ++j) acc[i][j] = vzero;

    const int nTiles = K / BKT;
    load_tiles(0, 0);
    ASYNC_WAIT();
    __syncthreads();

    for (int t = 0; t < nTiles; ++t) {
        const int cur = t & 1;
        if (t + 1 < nTiles) load_tiles(t + 1, cur ^ 1);

        // ---- 2 x (K=32) WMMA steps per LDS tile: 8 WMMAs per barrier ----
#pragma unroll
        for (int kc = 0; kc < 2; ++kc) {
            v16bf afrag[2], bfrag[2];
            const int kb = hl * 8;
#pragma unroll
            for (int tt = 0; tt < 2; ++tt) {
                const bf16* ap = &As[cur][wm0 + tt * 16 + l16][kc * 32 + hl * 16];
#pragma unroll
                for (int e = 0; e < 16; ++e)
                    afrag[tt][e] = ap[e];
                const bf16* bp = &Bs[cur][wn0 + tt * 16 + l16][kc * 32 + hl * 16];
#pragma unroll
                for (int e = 0; e < 16; ++e)
                    bfrag[tt][e] = bp[e];
            }
#pragma unroll
            for (int tm = 0; tm < 2; ++tm)
#pragma unroll
                for (int tn = 0; tn < 2; ++tn)
                    acc[tm][tn] = __builtin_amdgcn_wmma_f32_16x16x32_bf16(
                        false, afrag[tm], false, bfrag[tn], (short)0, acc[tm][tn], false, false);
        }

        ASYNC_WAIT();
        __syncthreads();
    }

    // ---- epilogue: scale + bias (+ residual add) ----
#pragma unroll
    for (int tn = 0; tn < 2; ++tn) {
        int gn = nBase + wn0 + tn * 16 + l16;
        if (gn >= N) continue;
        float bv = bias ? bias[gn] : 0.f;
#pragma unroll
        for (int tm = 0; tm < 2; ++tm) {
#pragma unroll
            for (int r = 0; r < 8; ++r) {
                int gm = mBase + wm0 + tm * 16 + hl * 8 + r;
                if (gm < M) {
                    size_t ci = (size_t)gm * ldc + gn;
                    float v = acc[tm][tn][r] * scale + bv;
                    if (flags & 1) Cg[ci] += v; else Cg[ci] = v;
                }
            }
        }
    }
}

// ---------------- elementwise / reduction kernels ----------------

__device__ inline float block_reduce_sum(float v, float* red, int tid) {
    red[tid] = v; __syncthreads();
    for (int s = 128; s; s >>= 1) { if (tid < s) red[tid] += red[tid + s]; __syncthreads(); }
    float r = red[0]; __syncthreads();
    return r;
}

__global__ __launch_bounds__(256) void embed_ln_kernel(
    const int* __restrict__ ids, const float* __restrict__ we, const float* __restrict__ pe,
    const float* __restrict__ sc, const float* __restrict__ bi, float* __restrict__ h)
{
    __shared__ float red[256];
    const int row = blockIdx.x, tid = threadIdx.x;
    const int id = ids[row];
    const int s  = row & (SS - 1);
    float v[3]; float sum = 0.f;
#pragma unroll
    for (int i = 0; i < 3; ++i) {
        int j = tid + i * 256;
        v[i] = we[(size_t)id * HH + j] + pe[(size_t)s * HH + j];
        sum += v[i];
    }
    float mu = block_reduce_sum(sum, red, tid) / (float)HH;
    float s2 = 0.f;
#pragma unroll
    for (int i = 0; i < 3; ++i) { float d = v[i] - mu; s2 += d * d; }
    float var = block_reduce_sum(s2, red, tid) / (float)HH;
    float rstd = rsqrtf(var + 1e-12f);
#pragma unroll
    for (int i = 0; i < 3; ++i) {
        int j = tid + i * 256;
        h[(size_t)row * HH + j] = (v[i] - mu) * rstd * sc[j] + bi[j];
    }
}

__global__ __launch_bounds__(256) void ln_bf16_kernel(
    const float* __restrict__ x, const float* __restrict__ sc, const float* __restrict__ bi,
    bf16* __restrict__ out)
{
    __shared__ float red[256];
    const int row = blockIdx.x, tid = threadIdx.x;
    const float* xp = x + (size_t)row * HH;
    float v[3]; float sum = 0.f;
#pragma unroll
    for (int i = 0; i < 3; ++i) { v[i] = xp[tid + i * 256]; sum += v[i]; }
    float mu = block_reduce_sum(sum, red, tid) / (float)HH;
    float s2 = 0.f;
#pragma unroll
    for (int i = 0; i < 3; ++i) { float d = v[i] - mu; s2 += d * d; }
    float var = block_reduce_sum(s2, red, tid) / (float)HH;
    float rstd = rsqrtf(var + 1e-12f);
#pragma unroll
    for (int i = 0; i < 3; ++i) {
        int j = tid + i * 256;
        out[(size_t)row * HH + j] = (bf16)((v[i] - mu) * rstd * sc[j] + bi[j]);
    }
}

__global__ __launch_bounds__(256) void gate_silu_kernel(
    const float* __restrict__ g, bf16* __restrict__ out)
{
    long i = (long)blockIdx.x * 256 + threadIdx.x;
    if (i >= (long)MT * HH) return;
    int m = (int)(i / HH), j = (int)(i % HH);
    const float* gp = g + (size_t)m * (3 * HH);
    float val = gp[j], gb = gp[HH + j], gc = gp[2 * HH + j];
    float silu = gb / (1.f + __expf(-gb));
    out[i] = (bf16)(gc * silu * val);
}

// Packs Q (rope), K (rope, [S,HD] rows), and V transposed ([HD,S] rows) per (b,head).
__global__ __launch_bounds__(256) void rope_pack_kernel(
    const float* __restrict__ qf, const float* __restrict__ kf, const float* __restrict__ vf,
    bf16* __restrict__ qbf, bf16* __restrict__ kbf, bf16* __restrict__ vbfT)
{
    long idx = (long)blockIdx.x * 256 + threadIdx.x;
    if (idx >= (long)BB * NHh * SS * HDh) return;
    int d  = (int)(idx & (HDh - 1));
    int s  = (int)((idx >> 6) & (SS - 1));
    int hh = (int)((idx >> 16) % NHh);
    int b  = (int)(idx / ((long)NHh * SS * HDh));
    int m  = b * SS + s;
    int kv = hh / (NHh / NKVh);
    int i  = d & 31;
    float freq = __expf(-(2.f * (float)i / (float)HDh) * 9.210340371976184f); // ln(10000)
    float ang = (float)s * freq;
    float cs = __cosf(ang), sn = __sinf(ang);
    const float* qp = qf + (size_t)m * HH + hh * HDh;
    const float* kp = kf + (size_t)m * (NKVh * HDh) + kv * HDh;
    float q1 = qp[i], q2 = qp[i + 32];
    float k1 = kp[i], k2 = kp[i + 32];
    float qo = (d < 32) ? (q1 * cs - q2 * sn) : (q1 * sn + q2 * cs);
    float ko = (d < 32) ? (k1 * cs - k2 * sn) : (k1 * sn + k2 * cs);
    qbf[idx] = (bf16)qo;
    kbf[idx] = (bf16)ko;
    // V^T: [b,h][HD][S]
    vbfT[(((size_t)(b * NHh + hh) * HDh + d) * SS) + s] =
        (bf16)vf[(size_t)m * (NKVh * HDh) + kv * HDh + d];
}

__global__ __launch_bounds__(256) void softmax_bf16_kernel(
    const float* __restrict__ scr, bf16* __restrict__ P)
{
    __shared__ float red[256];
    const int row = blockIdx.x, tid = threadIdx.x;
    const float* sp = scr + (size_t)row * SS;
    float v[4]; float mx = -3.4e38f;
#pragma unroll
    for (int i = 0; i < 4; ++i) { v[i] = sp[tid + i * 256]; mx = fmaxf(mx, v[i]); }
    red[tid] = mx; __syncthreads();
    for (int s = 128; s; s >>= 1) { if (tid < s) red[tid] = fmaxf(red[tid], red[tid + s]); __syncthreads(); }
    mx = red[0]; __syncthreads();
    float sum = 0.f;
#pragma unroll
    for (int i = 0; i < 4; ++i) { v[i] = __expf(v[i] - mx); sum += v[i]; }
    sum = block_reduce_sum(sum, red, tid);
    float inv = 1.f / sum;
#pragma unroll
    for (int i = 0; i < 4; ++i)
        P[(size_t)row * SS + tid + i * 256] = (bf16)(v[i] * inv);
}

__global__ __launch_bounds__(256) void cvt_bf16_kernel(
    const float* __restrict__ in, bf16* __restrict__ out, long n)
{
    for (long i = (long)blockIdx.x * 256 + threadIdx.x; i < n; i += (long)gridDim.x * 256)
        out[i] = (bf16)in[i];
}

// Transposing converter: in = NB batches of [R][C] f32 -> out = NB batches of [C][R] bf16.
__global__ __launch_bounds__(256) void cvtT_bf16_kernel(
    const float* __restrict__ in, bf16* __restrict__ out, int R, int C, long n)
{
    const long rc = (long)R * C;
    for (long idx = (long)blockIdx.x * 256 + threadIdx.x; idx < n; idx += (long)gridDim.x * 256) {
        long bch = idx / rc;
        long rem = idx - bch * rc;
        int r = (int)(rem / C);
        int c = (int)(rem - (long)r * C);
        out[bch * rc + (long)c * R + r] = (bf16)in[idx];
    }
}

// conv_w[c][o][i][k] -> wt[c][k][o][i]  (B-transposed operand: rows n=o, K-contig i)
__global__ __launch_bounds__(256) void convw_t_kernel(
    const float* __restrict__ w, bf16* __restrict__ wt, long n)
{
    for (long idx = (long)blockIdx.x * 256 + threadIdx.x; idx < n; idx += (long)gridDim.x * 256) {
        long k = idx % KK; long t = idx / KK;
        long i = t % HH;   t /= HH;
        long o = t % HH;   long c = t / HH;
        wt[(((c * KK + k) * HH + o) * HH) + i] = (bf16)w[idx];
    }
}

// ---------------- host side ----------------

static inline void launch_gemm(hipStream_t st,
    const bf16* A, const bf16* Bt, float* C, const float* bias,
    int M, int N, int K, int lda, int ldb, int ldc,
    int flags = 0, float scale = 1.f, int shift = 0, int seg = 0,
    int nz1 = 1, int nz2 = 1,
    long sA1 = 0, long sA2 = 0, long sB1 = 0, long sB2 = 0, long sC1 = 0, long sC2 = 0)
{
    dim3 grid((N + BN - 1) / BN, (M + BM - 1) / BM, nz1 * nz2);
    gemm_bf16_wmma<<<grid, 256, 0, st>>>(A, Bt, C, bias, M, N, K, lda, ldb, ldc,
                                         sA1, sA2, sB1, sB2, sC1, sC2, nz2,
                                         flags, scale, shift, seg);
}

static inline void launch_cvt(hipStream_t st, const float* in, bf16* out, long n) {
    cvt_bf16_kernel<<<2048, 256, 0, st>>>(in, out, n);
}
static inline void launch_cvtT(hipStream_t st, const float* in, bf16* out, int R, int C, long n) {
    cvtT_bf16_kernel<<<2048, 256, 0, st>>>(in, out, R, C, n);
}

extern "C" void kernel_launch(void* const* d_in, const int* in_sizes, int n_in,
                              void* d_out, int out_size, void* d_ws, size_t ws_size,
                              hipStream_t stream)
{
    const int*   ids       = (const int*)  d_in[0];
    const float* word_emb  = (const float*)d_in[1];
    const float* pos_emb   = (const float*)d_in[2];
    const float* emb_s     = (const float*)d_in[3];
    const float* emb_b     = (const float*)d_in[4];
    const float* c_norm_s  = (const float*)d_in[5];
    const float* c_norm_b  = (const float*)d_in[6];
    const float* gate_w    = (const float*)d_in[7];
    const float* gate_b    = (const float*)d_in[8];
    const float* conv_w    = (const float*)d_in[9];
    const float* conv_b    = (const float*)d_in[10];
    const float* out_w     = (const float*)d_in[11];
    const float* out_b     = (const float*)d_in[12];
    const float* a_norm_s  = (const float*)d_in[13];
    const float* a_norm_b  = (const float*)d_in[14];
    const float* q_w       = (const float*)d_in[15];
    const float* q_b       = (const float*)d_in[16];
    const float* k_w       = (const float*)d_in[17];
    const float* k_b       = (const float*)d_in[18];
    const float* v_w       = (const float*)d_in[19];
    const float* v_b       = (const float*)d_in[20];
    const float* o_w       = (const float*)d_in[21];
    const float* o_b       = (const float*)d_in[22];
    const float* lm_w      = (const float*)d_in[23];
    const float* lm_b      = (const float*)d_in[24];

    // ---- bump allocator over d_ws ----
    char* wp = (char*)d_ws;
    auto alloc = [&](size_t bytes) -> void* {
        void* p = (void*)wp;
        wp += (bytes + 255) & ~(size_t)255;
        return p;
    };

    const int KV = NKVh * HDh;                       // 256
    float* h      = (float*)alloc((size_t)MT * HH * 4);
    bf16*  xln    = (bf16*) alloc((size_t)MT * HH * 2);
    float* gbuf   = (float*)alloc((size_t)MT * 3 * HH * 4);
    bf16*  xg     = (bf16*) alloc((size_t)MT * HH * 2);
    float* ybuf   = (float*)alloc((size_t)MT * HH * 4);
    bf16*  ybf    = (bf16*) alloc((size_t)MT * HH * 2);
    float* qf     = (float*)alloc((size_t)MT * HH * 4);
    float* kf     = (float*)alloc((size_t)MT * KV * 4);
    float* vf     = (float*)alloc((size_t)MT * KV * 4);
    bf16*  qbf    = (bf16*) alloc((size_t)BB * NHh * SS * HDh * 2);
    bf16*  kbf    = (bf16*) alloc((size_t)BB * NHh * SS * HDh * 2);
    bf16*  vbfT   = (bf16*) alloc((size_t)BB * NHh * SS * HDh * 2);
    float* scores = (float*)alloc((size_t)BB * NHh * SS * SS * 4);
    bf16*  probs  = (bf16*) alloc((size_t)BB * NHh * SS * SS * 2);
    float* octx   = (float*)alloc((size_t)MT * HH * 4);
    bf16*  octxbf = (bf16*) alloc((size_t)MT * HH * 2);
    bf16*  hbf    = (bf16*) alloc((size_t)MT * HH * 2);
    // bf16 weights, all stored TRANSPOSED [N][K]
    bf16* gwb  = (bf16*)alloc((size_t)NCc * HH * 3 * HH * 2);
    bf16* cwt  = (bf16*)alloc((size_t)NCc * KK * HH * HH * 2);
    bf16* owb  = (bf16*)alloc((size_t)NCc * HH * HH * 2);
    bf16* qwb  = (bf16*)alloc((size_t)NAa * HH * HH * 2);
    bf16* kwb  = (bf16*)alloc((size_t)NAa * HH * KV * 2);
    bf16* vwb  = (bf16*)alloc((size_t)NAa * HH * KV * 2);
    bf16* o2wb = (bf16*)alloc((size_t)NAa * HH * HH * 2);
    bf16* lmwb = (bf16*)alloc((size_t)HH * VV * 2);

    // ---- weight conversion to transposed bf16 (deterministic, every call) ----
    launch_cvtT(stream, gate_w, gwb, HH, 3 * HH, (long)NCc * HH * 3 * HH);
    convw_t_kernel<<<2048, 256, 0, stream>>>(conv_w, cwt, (long)NCc * HH * HH * KK);
    launch_cvtT(stream, out_w, owb, HH, HH, (long)NCc * HH * HH);
    launch_cvtT(stream, q_w, qwb, HH, HH, (long)NAa * HH * HH);
    launch_cvtT(stream, k_w, kwb, HH, KV, (long)NAa * HH * KV);
    launch_cvtT(stream, v_w, vwb, HH, KV, (long)NAa * HH * KV);
    launch_cvtT(stream, o_w, o2wb, HH, HH, (long)NAa * HH * HH);
    launch_cvtT(stream, lm_w, lmwb, HH, VV, (long)HH * VV);

    // ---- embedding + LN ----
    embed_ln_kernel<<<MT, 256, 0, stream>>>(ids, word_emb, pos_emb, emb_s, emb_b, h);

    const char pat[16] = {'c','c','c','a','c','c','c','c','a','a','c','c','c','a','a','a'};
    int ci = 0, ai = 0;
    for (int bidx = 0; bidx < 16; ++bidx) {
        if (pat[bidx] == 'c') {
            // ---- conv block ----
            ln_bf16_kernel<<<MT, 256, 0, stream>>>(h, c_norm_s + (size_t)ci * HH,
                                                   c_norm_b + (size_t)ci * HH, xln);
            launch_gemm(stream, xln, gwb + (size_t)ci * HH * 3 * HH, gbuf,
                        gate_b + (size_t)ci * 3 * HH, MT, 3 * HH, HH, HH, HH, 3 * HH);
            gate_silu_kernel<<<(MT * HH + 255) / 256, 256, 0, stream>>>(gbuf, xg);
            for (int k = 0; k < KK; ++k) {
                launch_gemm(stream, xg, cwt + (((size_t)ci * KK + k) * HH * HH), ybuf,
                            (k == 0) ? (conv_b + (size_t)ci * HH) : nullptr,
                            MT, HH, HH, HH, HH, HH,
                            /*flags=*/(k == 0 ? 0 : 1), 1.f, /*shift=*/k - 1, /*seg=*/SS);
            }
            launch_cvt(stream, ybuf, ybf, (long)MT * HH);
            launch_gemm(stream, ybf, owb + (size_t)ci * HH * HH, h,
                        out_b + (size_t)ci * HH, MT, HH, HH, HH, HH, HH, /*flags=*/1);
            ++ci;
        } else {
            // ---- attention block ----
            ln_bf16_kernel<<<MT, 256, 0, stream>>>(h, a_norm_s + (size_t)ai * HH,
                                                   a_norm_b + (size_t)ai * HH, xln);
            launch_gemm(stream, xln, qwb + (size_t)ai * HH * HH, qf,
                        q_b + (size_t)ai * HH, MT, HH, HH, HH, HH, HH);
            launch_gemm(stream, xln, kwb + (size_t)ai * HH * KV, kf,
                        k_b + (size_t)ai * KV, MT, KV, HH, HH, HH, KV);
            launch_gemm(stream, xln, vwb + (size_t)ai * HH * KV, vf,
                        v_b + (size_t)ai * KV, MT, KV, HH, HH, HH, KV);
            {
                long n = (long)BB * NHh * SS * HDh;
                rope_pack_kernel<<<(int)((n + 255) / 256), 256, 0, stream>>>(qf, kf, vf, qbf, kbf, vbfT);
            }
            // scores[b,h] = Q[b,h] (S,HD) @ K[b,h]^T / sqrt(HD)   (kbf rows are already [n=S][k=HD])
            launch_gemm(stream, qbf, kbf, scores, nullptr,
                        SS, SS, HDh, HDh, HDh, SS,
                        /*flags=*/0, /*scale=*/0.125f, 0, 0,
                        /*nz1=*/BB, /*nz2=*/NHh,
                        (long)NHh * SS * HDh, (long)SS * HDh,
                        (long)NHh * SS * HDh, (long)SS * HDh,
                        (long)NHh * SS * SS,  (long)SS * SS);
            softmax_bf16_kernel<<<BB * NHh * SS, 256, 0, stream>>>(scores, probs);
            // octx[b,s,h*HD+d] = P[b,h] (S,S) @ V[b,h] (S,HD); V stored transposed [HD][S]
            launch_gemm(stream, probs, vbfT, octx, nullptr,
                        SS, HDh, SS, SS, SS, HH,
                        /*flags=*/0, 1.f, 0, 0,
                        /*nz1=*/BB, /*nz2=*/NHh,
                        (long)NHh * SS * SS, (long)SS * SS,
                        (long)NHh * HDh * SS, (long)HDh * SS,
                        (long)SS * HH, (long)HDh);
            launch_cvt(stream, octx, octxbf, (long)MT * HH);
            launch_gemm(stream, octxbf, o2wb + (size_t)ai * HH * HH, h,
                        o_b + (size_t)ai * HH, MT, HH, HH, HH, HH, HH, /*flags=*/1);
            ++ai;
        }
    }

    // ---- LM head: logits = h @ lm_w + lm_b  (lm_w stored transposed [V][H]) ----
    launch_cvt(stream, h, hbf, (long)MT * HH);
    launch_gemm(stream, hbf, lmwb, (float*)d_out, lm_b, MT, VV, HH, HH, HH, VV);
}
